// NTM_31525059953333
// MI455X (gfx1250) — matrix operand order repaired and checked by reading, not compiled, not run on hardware
//
#include <hip/hip_runtime.h>
#include <hip/hip_bf16.h>

#define B_ 64
#define T_ 32
#define I_ 512
#define H_ 1024
#define O_ 512
#define N_ 1024
#define M_ 64
#define NC  3852   // packed h-projection columns
#define NCP 3856   // padded to 16
#define EPS_ 1e-8f

typedef __bf16 bf16;
typedef __attribute__((ext_vector_type(16))) __bf16 v16bf;
typedef __attribute__((ext_vector_type(8)))  float  v8f;

union Frag { v16bf v; uint4 q[2]; };

__device__ __forceinline__ float sigmoidf_(float x) { return 1.f / (1.f + __expf(-x)); }
__device__ __forceinline__ float softplusf_(float x) { return (x > 20.f) ? x : log1pf(__expf(x)); }

// ---------------------------------------------------------------------------
// bf16 WMMA GEMM with async B-panel staging:
//   C[64 x Ncols] = A[64 x K](bf16,row-major) * Bpk + bias
// All 4 waves of a block share one col-tile, so the B panel (KS*1024 bytes,
// contiguous in the packed layout) is staged ONCE into LDS via
// global_load_async_to_lds_b128 (ASYNCcnt), then WMMA reads it via ds_load_b128.
// Bpk layout: per (coltile c, kstep ks) block of 512 halves:
//   dst[((c*KS+ks)*32 + lane)*16 + i] = W[n = c*16 + (lane&15)][k = ks*32 + (lane>>4)*16 + i]
// MODE 0: store fp32 C.   MODE 1: v = relu(x_t + relu(v+bias)) -> bf16 Obf (xi path).
// ---------------------------------------------------------------------------
template<int MODE>
__global__ void gemm_wmma(const bf16* __restrict__ A, int lda, int K,
                          const bf16* __restrict__ Bpk,
                          const float* __restrict__ bias, int Ncols,
                          float* __restrict__ C, int ldc,
                          const float* __restrict__ Xadd, int xstride,
                          bf16* __restrict__ Obf, int ldo)
{
    __shared__ bf16 sB[16384];                 // up to KS=32 panels (32 KB)

    const int lane = threadIdx.x & 31;
    const int rt   = threadIdx.x >> 5;         // row tile 0..3 (B=64)
    const int ct   = blockIdx.x;               // col tile
    const int KS   = K >> 5;

    // ---- async stage of this col-tile's B panel: global -> LDS ------------
    {
        const bf16* gsrc = Bpk + (((size_t)ct * KS) << 9);
        uint32_t lds_base = (uint32_t)(uintptr_t)(&sB[0]);
        const int total = KS << 9;             // halves in panel
        for (int ofs = threadIdx.x << 3; ofs < total; ofs += 128 * 8) {
            uint32_t ldst = lds_base + (uint32_t)(ofs << 1);
            unsigned long long ga = (unsigned long long)(uintptr_t)(gsrc + ofs);
            asm volatile("global_load_async_to_lds_b128 %0, %1, off"
                         :: "v"(ldst), "v"(ga) : "memory");
        }
        asm volatile("s_wait_asynccnt 0" ::: "memory");
        __syncthreads();
    }

    // ---- WMMA main loop: A from global, B from LDS ------------------------
    const int row = rt * 16 + (lane & 15);
    const bf16* ap  = A + (size_t)row * lda + ((lane >> 4) << 4);
    const bf16* lbp = sB + (lane << 4);

    v8f acc = {0.f, 0.f, 0.f, 0.f, 0.f, 0.f, 0.f, 0.f};
    for (int ks = 0; ks < KS; ++ks) {
        Frag a, b;
        a.q[0] = *(const uint4*)(ap);
        a.q[1] = *(const uint4*)(ap + 8);
        b.q[0] = *(const uint4*)(lbp);
        b.q[1] = *(const uint4*)(lbp + 8);
        acc = __builtin_amdgcn_wmma_f32_16x16x32_bf16(false, a.v, false, b.v,
                                                      (short)0, acc, false, false);
        ap  += 32;
        lbp += 512;
    }

    const int col = (ct << 4) + (lane & 15);
    const int rb  = rt * 16 + ((lane >> 4) << 3);
    const float bv = (col < Ncols) ? bias[col] : 0.f;
#pragma unroll
    for (int r = 0; r < 8; ++r) {
        float v = acc[r] + bv;
        const int rr = rb + r;
        if (MODE == 0) {
            C[(size_t)rr * ldc + col] = v;
        } else {
            v = fmaxf(v, 0.f);
            v = fmaxf(Xadd[(size_t)rr * xstride + col] + v, 0.f);
            Obf[(size_t)rr * ldo + col] = (bf16)v;
        }
    }
}

// Pack fp32 W (Nrows x K row-major, used as y = x @ W.T) into the WMMA B layout.
__global__ void pack_b(const float* __restrict__ W, int Nrows, int K,
                       bf16* __restrict__ dst, int Npad)
{
    const int KS = K >> 5;
    size_t total = ((size_t)(Npad >> 4)) * KS * 512;
    size_t idx = (size_t)blockIdx.x * blockDim.x + threadIdx.x;
    if (idx >= total) return;
    int i    = idx & 15;
    int lane = (idx >> 4) & 31;
    size_t blk = idx >> 9;
    int ks = (int)(blk % KS);
    int c  = (int)(blk / KS);
    int n = c * 16 + (lane & 15);
    int k = ks * 32 + ((lane >> 4) << 4) + i;
    float v = (n < Nrows) ? W[(size_t)n * K + k] : 0.f;
    dst[idx] = (bf16)v;
}

__global__ void h_to_bf16(const float* __restrict__ h, bf16* __restrict__ hbf)
{
    int idx = blockIdx.x * blockDim.x + threadIdx.x;   // B*H
    hbf[idx] = (bf16)h[idx];
}

// out(t-1) = sigmoid(Z[:, 3340:3852])
__global__ void emit_out(const float* __restrict__ Z, float* __restrict__ out, int tprev)
{
    int idx = blockIdx.x * blockDim.x + threadIdx.x;   // B*O
    int b = idx >> 9, j = idx & 511;
    out[((size_t)b * T_ + tprev) * O_ + j] = sigmoidf_(Z[(size_t)b * NCP + 3340 + j]);
}

// Per (head,b): k=tanh, ||k||, beta=softplus, g=sigmoid, s=softmax3, gamma=1+softplus
__global__ void head_params(const float* __restrict__ Z,
                            float* __restrict__ kbuf,    // [2][B][M]
                            float* __restrict__ prm)     // [2][B][8]
{
    int b = blockIdx.x & 63, head = blockIdx.x >> 6;
    int m = threadIdx.x;                                  // 64 threads
    __shared__ float red[64];
    const float* z = Z + (size_t)b * NCP;
    int off = head * 70;
    float kv = tanhf(z[off + m]);
    kbuf[((head * 64 + b) << 6) + m] = kv;
    red[m] = kv * kv;
    __syncthreads();
    for (int o = 32; o > 0; o >>= 1) { if (m < o) red[m] += red[m + o]; __syncthreads(); }
    if (m == 0) {
        float knorm = sqrtf(red[0]);
        float beta  = softplusf_(z[off + 64]);
        float g     = sigmoidf_(z[off + 65]);
        float s0 = z[off + 66], s1 = z[off + 67], s2 = z[off + 68];
        float mx = fmaxf(s0, fmaxf(s1, s2));
        float e0 = __expf(s0 - mx), e1 = __expf(s1 - mx), e2 = __expf(s2 - mx);
        float es = e0 + e1 + e2;
        float gamma = 1.f + softplusf_(z[off + 69]);
        float* p = prm + ((head * 64 + b) << 3);
        p[0] = beta; p[1] = g; p[2] = e0 / es; p[3] = e1 / es; p[4] = e2 / es;
        p[5] = gamma; p[6] = knorm;
    }
}

// One wave per (b,n): cosine-sim logits (beta applied) for both heads.
__global__ void sim_kernel(const float* __restrict__ mem,
                           const float* __restrict__ kbuf,
                           const float* __restrict__ prm,
                           float* __restrict__ simr, float* __restrict__ simw)
{
    int wid  = blockIdx.x * (blockDim.x >> 5) + (threadIdx.x >> 5);   // b*N + n
    int lane = threadIdx.x & 31;
    int b = wid >> 10;
    const float* row = mem + ((size_t)wid << 6);
    float v0 = row[lane], v1 = row[lane + 32];
    float kr0 = kbuf[((0 * 64 + b) << 6) + lane], kr1 = kbuf[((0 * 64 + b) << 6) + lane + 32];
    float kw0 = kbuf[((1 * 64 + b) << 6) + lane], kw1 = kbuf[((1 * 64 + b) << 6) + lane + 32];
    float dr = v0 * kr0 + v1 * kr1;
    float dw = v0 * kw0 + v1 * kw1;
    float nn = v0 * v0 + v1 * v1;
    for (int o = 16; o > 0; o >>= 1) {
        dr += __shfl_xor(dr, o, 32);
        dw += __shfl_xor(dw, o, 32);
        nn += __shfl_xor(nn, o, 32);
    }
    if (lane == 0) {
        float mn = sqrtf(nn);
        const float* pr = prm + ((0 * 64 + b) << 3);
        const float* pw = prm + ((1 * 64 + b) << 3);
        simr[wid] = pr[0] * dr / (mn * pr[6] + EPS_);
        simw[wid] = pw[0] * dw / (mn * pw[6] + EPS_);
    }
}

__device__ __forceinline__ float blk_max(float v, float* s) {
    int t = threadIdx.x;
    s[t] = v; __syncthreads();
    for (int o = 128; o > 0; o >>= 1) { if (t < o) s[t] = fmaxf(s[t], s[t + o]); __syncthreads(); }
    float r = s[0]; __syncthreads(); return r;
}
__device__ __forceinline__ float blk_sum(float v, float* s) {
    int t = threadIdx.x;
    s[t] = v; __syncthreads();
    for (int o = 128; o > 0; o >>= 1) { if (t < o) s[t] += s[t + o]; __syncthreads(); }
    float r = s[0]; __syncthreads(); return r;
}

// softmax + interpolate + circular shift + sharpen + renorm; in-place on w buffer.
__global__ void addr_finalize(const float* __restrict__ simr, const float* __restrict__ simw,
                              const float* __restrict__ prm,
                              float* __restrict__ wrbuf, float* __restrict__ wwbuf)
{
    int b = blockIdx.x & 63, head = blockIdx.x >> 6;
    const float* sim = (head ? simw : simr) + ((size_t)b << 10);
    float* wbuf = (head ? wwbuf : wrbuf) + ((size_t)b << 10);
    const float* p = prm + ((head * 64 + b) << 3);
    float g = p[1], s0 = p[2], s1 = p[3], s2 = p[4], gamma = p[5];

    __shared__ float wg[1024];
    __shared__ float sred[256];
    int t = threadIdx.x;

    float lv[4];
    float mx = -1e30f;
#pragma unroll
    for (int j = 0; j < 4; ++j) { lv[j] = sim[t + j * 256]; mx = fmaxf(mx, lv[j]); }
    mx = blk_max(mx, sred);
    float ls = 0.f;
#pragma unroll
    for (int j = 0; j < 4; ++j) { lv[j] = __expf(lv[j] - mx); ls += lv[j]; }
    ls = blk_sum(ls, sred);
    float inv = 1.f / ls;
#pragma unroll
    for (int j = 0; j < 4; ++j) {
        int n = t + j * 256;
        wg[n] = g * (lv[j] * inv) + (1.f - g) * wbuf[n];
    }
    __syncthreads();
    float wv[4]; float wsum = 0.f;
#pragma unroll
    for (int j = 0; j < 4; ++j) {
        int n = t + j * 256;
        float wt = s0 * wg[(n + 1023) & 1023] + s1 * wg[n] + s2 * wg[(n + 1) & 1023];
        float w = __expf(gamma * __logf(wt + 1e-12f));
        wv[j] = w; wsum += w;
    }
    wsum = blk_sum(wsum, sred);
    float winv = 1.f / (wsum + EPS_);
#pragma unroll
    for (int j = 0; j < 4; ++j) wbuf[t + j * 256] = wv[j] * winv;
}

// r = wr . mem (old mem); mem = mem*(1 - ww*e) + ww*a. One block per batch.
__global__ void read_write_mem(float* __restrict__ mem,
                               const float* __restrict__ wr, const float* __restrict__ ww,
                               const float* __restrict__ Z, bf16* __restrict__ rbf)
{
    int b = blockIdx.x;
    int t = threadIdx.x;
    int m = t & 63, sub = t >> 6;
    __shared__ float e_s[64], a_s[64], red[256];
    if (t < 64) {
        const float* z = Z + (size_t)b * NCP;
        e_s[t] = sigmoidf_(z[140 + t]);
        a_s[t] = tanhf(z[204 + t]);
    }
    __syncthreads();
    float em = e_s[m], am = a_s[m];
    float racc = 0.f;
    float* mb = mem + ((size_t)b << 16);
    const float* wrb = wr + ((size_t)b << 10);
    const float* wwb = ww + ((size_t)b << 10);
    for (int n = sub; n < 1024; n += 4) {
        float wrn = wrb[n], wwn = wwb[n];
        float v = mb[(n << 6) + m];
        racc += wrn * v;
        mb[(n << 6) + m] = v * (1.f - wwn * em) + wwn * am;
    }
    red[t] = racc;
    __syncthreads();
    if (sub == 0) {
        float r = red[m] + red[64 + m] + red[128 + m] + red[192 + m];
        rbf[(b << 6) + m] = (bf16)r;
    }
}

// GRU gate combine; also emits next step's bf16 copy of h (saves a launch).
__global__ void gru_combine(const float* __restrict__ gi, const float* __restrict__ Z,
                            float* __restrict__ h, bf16* __restrict__ hbf)
{
    int idx = blockIdx.x * blockDim.x + threadIdx.x;   // B*H
    int b = idx >> 10, j = idx & 1023;
    const float* g = gi + ((size_t)b * 3072);
    const float* z = Z + (size_t)b * NCP + 268;
    float r  = sigmoidf_(g[j] + z[j]);
    float zt = sigmoidf_(g[j + 1024] + z[j + 1024]);
    float n  = tanhf(g[j + 2048] + r * z[j + 2048]);
    float hv = (1.f - zt) * n + zt * h[idx];
    h[idx] = hv;
    hbf[idx] = (bf16)hv;
}

// ---------------------------------------------------------------------------
extern "C" void kernel_launch(void* const* d_in, const int* in_sizes, int n_in,
                              void* d_out, int out_size, void* d_ws, size_t ws_size,
                              hipStream_t stream)
{
    (void)in_sizes; (void)n_in; (void)out_size; (void)ws_size;
    const float* x     = (const float*)d_in[0];
    const float* b_r2i = (const float*)d_in[30];
    const float* bih   = (const float*)d_in[32];

    // ---- output regions double as recurrent state ----
    float* out = (float*)d_out;
    float* hst = out + (size_t)B_ * T_ * O_;
    float* wr  = hst + (size_t)B_ * H_;
    float* ww  = wr  + (size_t)B_ * N_;
    float* mem = ww  + (size_t)B_ * N_;

    // ---- workspace carve ----
    char* wp = (char*)d_ws;
    auto carve = [&](size_t bytes) { char* p = wp; wp += (bytes + 255) & ~(size_t)255; return p; };
    float* Wcat_f32 = (float*)carve((size_t)NC * H_ * 4);
    bf16*  Wcat_pk  = (bf16*) carve((size_t)(NCP/16) * (H_/32) * 512 * 2);
    float* bcat     = (float*)carve((size_t)NCP * 4);
    bf16*  Wr2i_pk  = (bf16*) carve((size_t)(I_/16) * (M_/32) * 512 * 2);
    bf16*  Wih_pk   = (bf16*) carve((size_t)(3*H_/16) * (I_/32) * 512 * 2);
    float* Z        = (float*)carve((size_t)B_ * NCP * 4);
    bf16*  hbf      = (bf16*) carve((size_t)B_ * H_ * 2);
    float* kbuf     = (float*)carve((size_t)2 * B_ * M_ * 4);
    float* prm      = (float*)carve((size_t)2 * B_ * 8 * 4);
    float* simr     = (float*)carve((size_t)B_ * N_ * 4);
    float* simw     = (float*)carve((size_t)B_ * N_ * 4);
    bf16*  rbf      = (bf16*) carve((size_t)B_ * M_ * 2);
    bf16*  xibf     = (bf16*) carve((size_t)B_ * I_ * 2);
    float* gi       = (float*)carve((size_t)B_ * 3 * H_ * 4);

    // ---- concatenate h-projection weights/biases (row-concat = contiguous D2D) ----
    static const int widx[14] = {5,7,9,11,13,15,17,19,21,23,25,27,33,35};
    static const int bidx[14] = {6,8,10,12,14,16,18,20,22,24,26,28,34,36};
    static const int rows_[14] = {64,1,1,3,1,64,1,1,3,1,64,64,3072,512};
    size_t off = 0;
    for (int i = 0; i < 14; ++i) {
        hipMemcpyAsync(Wcat_f32 + off * H_, d_in[widx[i]],
                       (size_t)rows_[i] * H_ * sizeof(float), hipMemcpyDeviceToDevice, stream);
        hipMemcpyAsync(bcat + off, d_in[bidx[i]],
                       (size_t)rows_[i] * sizeof(float), hipMemcpyDeviceToDevice, stream);
        off += rows_[i];
    }

    // ---- init recurrent state from inputs (never mutate d_in) ----
    hipMemcpyAsync(hst, d_in[1], (size_t)B_ * H_ * 4, hipMemcpyDeviceToDevice, stream);
    hipMemcpyAsync(wr,  d_in[2], (size_t)B_ * N_ * 4, hipMemcpyDeviceToDevice, stream);
    hipMemcpyAsync(ww,  d_in[3], (size_t)B_ * N_ * 4, hipMemcpyDeviceToDevice, stream);
    hipMemcpyAsync(mem, d_in[4], (size_t)B_ * N_ * M_ * 4, hipMemcpyDeviceToDevice, stream);

    // ---- pack weights into WMMA B layout (bf16) ----
    {
        size_t tot = (size_t)(NCP/16) * (H_/32) * 512;
        pack_b<<<(int)((tot + 255) / 256), 256, 0, stream>>>(Wcat_f32, NC, H_, Wcat_pk, NCP);
        tot = (size_t)(I_/16) * (M_/32) * 512;
        pack_b<<<(int)((tot + 255) / 256), 256, 0, stream>>>((const float*)d_in[29], I_, M_, Wr2i_pk, I_);
        tot = (size_t)(3*H_/16) * (I_/32) * 512;
        pack_b<<<(int)((tot + 255) / 256), 256, 0, stream>>>((const float*)d_in[31], 3*H_, I_, Wih_pk, 3*H_);
    }

    // initial h -> bf16 (later steps produce hbf inside gru_combine)
    h_to_bf16<<<(B_*H_)/256, 256, 0, stream>>>(hst, hbf);

    // ---- time loop (iteration T_ only produces out[T_-1]) ----
    for (int t = 0; t <= T_; ++t) {
        gemm_wmma<0><<<NCP/16, 128, 0, stream>>>(hbf, H_, H_, Wcat_pk, bcat, NC,
                                                 Z, NCP, (const float*)nullptr, 0,
                                                 (bf16*)nullptr, 0);
        if (t > 0)
            emit_out<<<(B_*O_)/256, 256, 0, stream>>>(Z, out, t - 1);
        if (t == T_) break;

        head_params<<<2*B_, 64, 0, stream>>>(Z, kbuf, prm);
        sim_kernel<<<(B_*N_)/8, 256, 0, stream>>>(mem, kbuf, prm, simr, simw);
        addr_finalize<<<2*B_, 256, 0, stream>>>(simr, simw, prm, wr, ww);
        read_write_mem<<<B_, 256, 0, stream>>>(mem, wr, ww, Z, rbf);

        gemm_wmma<1><<<I_/16, 128, 0, stream>>>(rbf, M_, M_, Wr2i_pk, b_r2i, I_,
                                                (float*)nullptr, 0,
                                                x + (size_t)t * I_, T_ * I_, xibf, I_);
        gemm_wmma<0><<<(3*H_)/16, 128, 0, stream>>>(xibf, I_, I_, Wih_pk, bih, 3*H_,
                                                    gi, 3*H_, (const float*)nullptr, 0,
                                                    (bf16*)nullptr, 0);
        gru_combine<<<(B_*H_)/256, 256, 0, stream>>>(gi, Z, hst, hbf);
    }
}
